// ContextPatchEmbed_29222957482773
// MI455X (gfx1250) — matrix operand order repaired and checked by reading, not compiled, not run on hardware
//
#include <hip/hip_runtime.h>
#include <hip/hip_bf16.h>
#include <stdint.h>

// ---------------------------------------------------------------------------
// ContextPatchEmbed for MI455X (gfx1250, wave32, WMMA)
// B=32 C=8 H=W=224 E=768 P=16  ->  out (B, C*196, 768) f32
// Pipeline exploits linearity: ctx is computed from the mean patch BEFORE the
// big GEMM, so the 50176x768x256 patchify GEMM runs once, fused with
// (+proj_b +ctx), writing the final layout directly. ~256 MB total traffic.
// ---------------------------------------------------------------------------

typedef __attribute__((ext_vector_type(16))) __bf16       v16bf;
typedef __attribute__((ext_vector_type(8)))  float        v8f;
typedef __attribute__((ext_vector_type(8)))  unsigned int v8u;

#define E_DIM   768
#define KP      256        // P*P
#define HPD     14         // H/P
#define NPATCH  196        // 14*14
#define BC      256        // B*C
#define LDA     264        // LDS A row stride in halves (bank-conflict padding)
#define NSEG    32

static __device__ __forceinline__ unsigned short f2bf(float f) {
  union { float f; unsigned u; } v; v.f = f;
  unsigned r = v.u + 0x7FFFu + ((v.u >> 16) & 1u);   // round-to-nearest-even
  return (unsigned short)(r >> 16);
}

static __device__ __forceinline__ v16bf frag_from(const uint4 lo, const uint4 hi) {
  v8u t;
  t[0] = lo.x; t[1] = lo.y; t[2] = lo.z; t[3] = lo.w;
  t[4] = hi.x; t[5] = hi.y; t[6] = hi.z; t[7] = hi.w;
  return __builtin_bit_cast(v16bf, t);
}

// K0: proj_w f32 (E,P,P) -> bf16 W[e][k], k = p*16+q (same flat order)
__global__ void k0_convert_w(const float* __restrict__ pw,
                             unsigned short* __restrict__ wbf) {
  int i = blockIdx.x * 256 + threadIdx.x;
  if (i < E_DIM * KP) wbf[i] = f2bf(pw[i]);
}

// K1: mean patch per (b,c): mp[bc][k] = mean over (hp,wp) of x patches (f32)
__global__ void k1_mean_patch(const float* __restrict__ x,
                              float* __restrict__ mp) {
  const int bc = blockIdx.x;
  const int k  = threadIdx.x;          // 0..255
  const int p  = k >> 4, q = k & 15;
  const float* xb = x + (size_t)bc * (224 * 224);
  float s = 0.f;
  for (int hp = 0; hp < HPD; ++hp)
    for (int wp = 0; wp < HPD; ++wp)
      s += xb[(hp * 16 + p) * 224 + wp * 16 + q];
  mp[bc * KP + k] = s * (1.0f / 196.0f);
}

// K2: av_patch[bc][e] = dot(mp[bc], proj_w[e]) + proj_b[e]  (f32, linearity)
__global__ void k2_av_patch(const float* __restrict__ mp,
                            const float* __restrict__ pw,
                            const float* __restrict__ pb,
                            float* __restrict__ av) {
  __shared__ float mrow[KP];
  const int bc = blockIdx.x;
  mrow[threadIdx.x] = mp[bc * KP + threadIdx.x];
  __syncthreads();
  for (int e = threadIdx.x; e < E_DIM; e += 256) {
    const float* wr = pw + (size_t)e * KP;
    float acc = 0.f;
    #pragma unroll 8
    for (int k = 0; k < KP; ++k) acc += mrow[k] * wr[k];
    av[bc * E_DIM + e] = acc + pb[e];
  }
}

// K3: segment mean over batch by channel id (32 segments)
__global__ void k3_segment_mean(const float* __restrict__ av,
                                const int* __restrict__ seg,
                                float* __restrict__ means) {
  const int s   = blockIdx.x;          // 0..31
  const int tid = threadIdx.x;         // 0..255 -> e, e+256, e+512
  float a0 = 0.f, a1 = 0.f, a2 = 0.f;
  int cnt = 0;
  for (int bc = 0; bc < BC; ++bc) {
    if (seg[bc] == s) {
      ++cnt;
      const float* r = av + bc * E_DIM;
      a0 += r[tid]; a1 += r[tid + 256]; a2 += r[tid + 512];
    }
  }
  const float inv = 1.0f / fmaxf((float)cnt, 1.0f);
  float* m = means + s * E_DIM;
  m[tid] = a0 * inv; m[tid + 256] = a1 * inv; m[tid + 512] = a2 * inv;
}

// K4: ctx[bc][e'] = dot(means[seg[bc]], ctx_w[e']) + ctx_b[e']
__global__ void k4_ctx(const float* __restrict__ means,
                       const int* __restrict__ seg,
                       const float* __restrict__ cw,
                       const float* __restrict__ cb,
                       float* __restrict__ ctx) {
  __shared__ float m[E_DIM];
  const int bc = blockIdx.x;
  const int s  = seg[bc];
  for (int e = threadIdx.x; e < E_DIM; e += 256) m[e] = means[s * E_DIM + e];
  __syncthreads();
  for (int ep = threadIdx.x; ep < E_DIM; ep += 256) {
    const float* wr = cw + (size_t)ep * E_DIM;
    float acc = cb[ep];
    #pragma unroll 8
    for (int k = 0; k < E_DIM; ++k) acc += m[k] * wr[k];
    ctx[bc * E_DIM + ep] = acc;
  }
}

// K5: main patchify GEMM (bf16 WMMA, f32 acc) fused with +proj_b +ctx, writes
// final output layout directly. Grid (256 bc, 2 n-splits) x 256 threads.
__global__ void __launch_bounds__(256)
k5_patch_gemm(const float* __restrict__ x,
              const unsigned short* __restrict__ wbf,   // bf16 W[e][k]
              const float* __restrict__ pb,
              const float* __restrict__ ctx,
              float* __restrict__ out) {
  extern __shared__ unsigned short As[];                // 208 x LDA halves
  const int bc    = blockIdx.x;
  const int split = blockIdx.y;                         // 0,1 -> N halves
  const int tid   = threadIdx.x;
  const int lane  = tid & 31;
  const int wv    = tid >> 5;                           // 8 waves
  const int nhalf = lane >> 4;                          // 0 | 1
  const int ncol  = lane & 15;

  // Stage x patches -> LDS bf16, patch-major rows: As[m][k], m=hp*14+wp.
  // Wave wv owns image rows [wv*28, wv*28+28); lanes stride columns
  // (coalesced 128B global reads, shifts/masks only - no integer division).
  const float* xb = x + (size_t)bc * (224 * 224);
  for (int h = wv * 28; h < wv * 28 + 28; ++h) {
    const int mrow = (h >> 4) * HPD;                    // hp*14
    const int krow = (h & 15) << 4;                     // p*16
    const float* xr = xb + h * 224;
    #pragma unroll
    for (int i = 0; i < 7; ++i) {
      const int w = lane + i * 32;
      As[(mrow + (w >> 4)) * LDA + (krow | (w & 15))] = f2bf(xr[w]);
    }
  }
  // zero pad rows 196..207 (read by last M-tile, never stored)
  for (int idx = tid; idx < 12 * LDA; idx += 256)
    As[NPATCH * LDA + idx] = 0;
  __syncthreads();

  // Each wave: 3 N-tiles of 16 cols; 13 M-tiles of 16 rows; K = 8 x 32
  for (int nt = split * 24 + wv; nt < split * 24 + 24; nt += 8) {
    const int e = nt * 16 + ncol;                       // this lane's column

    // Preload all 8 B fragments (same k order as A: lanes<16 K[0..16), lanes>=16 K[16..32))
    v16bf bfrag[8];
    const unsigned short* wrow = wbf + (size_t)e * KP;
    #pragma unroll
    for (int kk = 0; kk < 8; ++kk) {
      const uint4* p = reinterpret_cast<const uint4*>(wrow + kk * 32 + nhalf * 16);
      bfrag[kk] = frag_from(p[0], p[1]);
    }
    // Prefetch next N-tile's weight row (global_prefetch_b8) while computing.
    if (nt + 8 < 48)
      __builtin_prefetch(wbf + (size_t)(e + 128) * KP, 0, 3);

    // bias + ctx folded into the WMMA accumulator init: D = A*B + C
    const float addv = pb[e] + ctx[bc * E_DIM + e];
    v8f cinit;
    #pragma unroll
    for (int v = 0; v < 8; ++v) cinit[v] = addv;

    for (int mt = 0; mt < 13; ++mt) {
      v8f c = cinit;
      const int row = mt * 16 + ncol;                   // A row = M = lane%16
      #pragma unroll
      for (int kk = 0; kk < 8; ++kk) {
        // A frag: lane<16 -> K[0..16); lane>=16 -> K[16..32)
        const unsigned short* arow = As + row * LDA + kk * 32 + nhalf * 16;
        const v16bf a = frag_from(*reinterpret_cast<const uint4*>(arow),
                                  *reinterpret_cast<const uint4*>(arow + 8));
        c = __builtin_amdgcn_wmma_f32_16x16x32_bf16(
                false, a, false, bfrag[kk], (short)0, c, false, false);
      }
      // C layout: VGPR v -> M = v + 8*nhalf, N = lane%16
      #pragma unroll
      for (int v = 0; v < 8; ++v) {
        const int m = mt * 16 + v + (nhalf << 3);
        if (m < NPATCH)
          out[((size_t)bc * NPATCH + m) * E_DIM + e] = c[v];
      }
    }
  }
}

// ---------------------------------------------------------------------------
extern "C" void kernel_launch(void* const* d_in, const int* in_sizes, int n_in,
                              void* d_out, int out_size, void* d_ws, size_t ws_size,
                              hipStream_t stream) {
  (void)in_sizes; (void)n_in; (void)out_size; (void)ws_size;
  const float* x   = (const float*)d_in[0];
  const int*   seg = (const int*)d_in[1];     // channel_idx (int on device)
  const float* pw  = (const float*)d_in[2];   // proj_w (E,P,P)
  const float* pb  = (const float*)d_in[3];   // proj_b (E)
  const float* cw  = (const float*)d_in[4];   // ctx_w (E,E)
  const float* cb  = (const float*)d_in[5];   // ctx_b (E)
  float* out = (float*)d_out;

  // workspace layout (bytes): ~2.33 MB total
  char* ws = (char*)d_ws;
  unsigned short* wbf   = (unsigned short*)(ws);            // 768*256*2 = 393216
  float*          mp    = (float*)(ws + 393216);            // 256*256*4 = 262144
  float*          av    = (float*)(ws + 655360);            // 256*768*4 = 786432
  float*          means = (float*)(ws + 1441792);           // 32*768*4  =  98304
  float*          ctx   = (float*)(ws + 1540096);           // 256*768*4 = 786432

  k0_convert_w   <<<768, 256, 0, stream>>>(pw, wbf);
  k1_mean_patch  <<<BC,  256, 0, stream>>>(x, mp);
  k2_av_patch    <<<BC,  256, 0, stream>>>(mp, pw, pb, av);
  k3_segment_mean<<<NSEG, 256, 0, stream>>>(av, seg, means);
  k4_ctx         <<<BC,  256, 0, stream>>>(means, seg, cw, cb, ctx);

  dim3 g5(BC, 2);
  const size_t ldsBytes = 208 * LDA * sizeof(unsigned short);   // 109,824 B
  k5_patch_gemm<<<g5, 256, ldsBytes, stream>>>(x, wbf, pb, ctx, out);
}
